// CLASS_NN_Embed_cluster_57097295233114
// MI455X (gfx1250) — compile-verified, matrix operand order checked by reading, not run on hardware
//
#include <hip/hip_runtime.h>
#include <hip/hip_bf16.h>

typedef __attribute__((ext_vector_type(16))) __bf16 v16bf;
typedef __attribute__((ext_vector_type(8)))  float  v8f;

#define NROWS 4096
#define COLS  40
#define EMB   64
#define VALS  64
#define DCAT  2816
#define NC    100

// ---------------------------------------------------------------------------
// WMMA fragment loader: 16-bit 16x32 fragment from LDS row-major rows.
// lane L: row = caller-selected (lane&15), K-half = lane>>4.
// Layout makes each fragment two contiguous 16B chunks -> 2x ds_load_b128.
// Requires 16B-aligned rows (row stride multiple of 8 halves).
// ---------------------------------------------------------------------------
__device__ __forceinline__ v16bf ld_frag(const __bf16* row, int k0, int lh) {
  union { v16bf v; uint4 q[2]; } r;
  const __bf16* p = row + k0 + 8 * lh;
  r.q[0] = *(const uint4*)(p);
  r.q[1] = *(const uint4*)(p + 16);
  return r.v;
}

__device__ __forceinline__ v8f wmma_bf16(v16bf a, v16bf b, v8f c) {
  return __builtin_amdgcn_wmma_f32_16x16x32_bf16(false, a, false, b, (short)0, c,
                                                 false, false);
}

// gfx1250 async copy: 16B global -> LDS, tracked by ASYNCcnt.
// LDS byte address = low 32 bits of the flat pointer (ISA flat->LDS aperture).
__device__ __forceinline__ void async_cp16(void* lds, const void* gptr) {
  unsigned l = (unsigned)(size_t)lds;
  asm volatile("global_load_async_to_lds_b128 %0, %1, off"
               :: "v"(l), "v"(gptr) : "memory");
}
// allow the 3 newest async copies (next tile) to stay in flight
__device__ __forceinline__ void async_wait3() {
  asm volatile("s_wait_asynccnt 0x3" ::: "memory");
}
__device__ __forceinline__ void async_wait0() {
  asm volatile("s_wait_asynccnt 0x0" ::: "memory");
}

// ---------------------------------------------------------------------------
// GEMM: C[M,N] = act(A[row_map[m]] (bf16,lda) @ Wt^T + bias)
//   Wt is the bf16 transposed weight, Wt[n][k], row-major [N,K].
// Block tile 128x64, BK=32, 8 waves (wave32), each wave 32x32.
// Double-buffered LDS tiles filled by GLOBAL_LOAD_ASYNC_TO_LDS_B128;
// s_wait_asynccnt 0x3 overlaps next-tile copy with current-tile WMMA.
// Assumes: M % 128 == 0, K % 32 == 0, lda/K row strides 16B-aligned.
// ---------------------------------------------------------------------------
#define BM 128
#define BN 64
#define BK 32
#define BKP 40  // padded row stride (halves): 80B, 16B-aligned

__global__ void __launch_bounds__(256)
gemm_bf16_kernel(const __bf16* __restrict__ A, int lda,
                 const int* __restrict__ row_map,
                 const __bf16* __restrict__ Wt, const float* __restrict__ bias,
                 int M, int N, int K, int relu,
                 float* __restrict__ Cf, __bf16* __restrict__ Cb, int ldc) {
  __shared__ __attribute__((aligned(16))) __bf16 As[2][BM][BKP];
  __shared__ __attribute__((aligned(16))) __bf16 Bs[2][BN][BKP];

  const int t    = threadIdx.x;
  const int lane = t & 31;
  const int wave = t >> 5;
  const int lm   = lane & 15;
  const int lh   = lane >> 4;
  const int m_blk = blockIdx.y * BM;
  const int n_blk = blockIdx.x * BN;
  const int wm = (wave & 3) * 32;
  const int wn = (wave >> 2) * 32;

  // --- staging assignments (K-invariant, hoisted) -------------------------
  // A tile: 128 rows x 32 halves = 512 x 16B chunks; thread t owns chunks
  // t and t+256: row = chunk>>2, segment = (chunk&3)*8 halves.
  const int am0 = t >> 2, aseg = (t & 3) * 8;
  const int am1 = am0 + 64;
  const int gm0 = m_blk + am0, gm1 = m_blk + am1;
  const __bf16* arow0 = A + (size_t)(row_map ? row_map[gm0] : gm0) * lda;
  const __bf16* arow1 = A + (size_t)(row_map ? row_map[gm1] : gm1) * lda;
  // B tile: 64 rows x 4 chunks = 256; one per thread. Source row clamped so
  // every wave issues exactly 3 asyncs per tile (uniform ASYNCcnt); rows with
  // n >= N hold junk that only feeds masked-out output columns.
  const int bn = t >> 2, bseg = (t & 3) * 8;
  const int bnc = (n_blk + bn < N) ? (n_blk + bn) : (N - 1);
  const __bf16* brow = Wt + (size_t)bnc * K;

  v8f acc[2][2];
#pragma unroll
  for (int i = 0; i < 2; ++i)
#pragma unroll
    for (int j = 0; j < 2; ++j)
      acc[i][j] = (v8f){0.f, 0.f, 0.f, 0.f, 0.f, 0.f, 0.f, 0.f};

  auto stage = [&](int b, int k0) {
    async_cp16(&As[b][am0][aseg], arow0 + k0 + aseg);
    async_cp16(&As[b][am1][aseg], arow1 + k0 + aseg);
    async_cp16(&Bs[b][bn][bseg], brow + k0 + bseg);
  };

  stage(0, 0);
  int buf = 0;
  for (int k0 = 0; k0 < K; k0 += BK) {
    if (k0 + BK < K) {
      stage(buf ^ 1, k0 + BK);  // prefetch next tile into the other buffer
      async_wait3();            // current tile resident (in-order completion)
    } else {
      async_wait0();
    }
    __syncthreads();

    v16bf a0 = ld_frag(&As[buf][wm + lm][0], 0, lh);
    v16bf a1 = ld_frag(&As[buf][wm + 16 + lm][0], 0, lh);
    v16bf b0 = ld_frag(&Bs[buf][wn + lm][0], 0, lh);
    v16bf b1 = ld_frag(&Bs[buf][wn + 16 + lm][0], 0, lh);
    acc[0][0] = wmma_bf16(a0, b0, acc[0][0]);
    acc[0][1] = wmma_bf16(a0, b1, acc[0][1]);
    acc[1][0] = wmma_bf16(a1, b0, acc[1][0]);
    acc[1][1] = wmma_bf16(a1, b1, acc[1][1]);
    __syncthreads();
    buf ^= 1;
  }

  // epilogue: C/D layout m = r + 8*lh, n = lm
#pragma unroll
  for (int i = 0; i < 2; ++i)
#pragma unroll
    for (int j = 0; j < 2; ++j) {
      int n = n_blk + wn + 16 * j + lm;
      if (n < N) {
        float bv = bias ? bias[n] : 0.f;
#pragma unroll
        for (int r = 0; r < 8; ++r) {
          int m = m_blk + wm + 16 * i + r + 8 * lh;
          if (m < M) {
            float v = acc[i][j][r] + bv;
            if (relu) v = fmaxf(v, 0.f);
            if (Cf) Cf[(size_t)m * ldc + n] = v;
            if (Cb) Cb[(size_t)m * ldc + n] = (__bf16)v;
          }
        }
      }
    }
}

// ---------------------------------------------------------------------------
// Weight prep: Wt[n*K + k] = (bf16)W[k*N + n]
// ---------------------------------------------------------------------------
__global__ void twt_kernel(const float* __restrict__ W, __bf16* __restrict__ Wt,
                           int K, int N) {
  int i = blockIdx.x * blockDim.x + threadIdx.x;
  if (i < K * N) {
    int k = i / N, n = i % N;
    Wt[(size_t)n * K + k] = (__bf16)W[i];
  }
}

// ---------------------------------------------------------------------------
// Per-row attention: scores = q k^T (WMMA), softmax, ctx = probs v (WMMA),
// LayerNorm over flattened 2560. One block (4 waves) per batch row.
// ---------------------------------------------------------------------------
__global__ void __launch_bounds__(128)
attn_kernel(const __bf16* __restrict__ Q, const __bf16* __restrict__ Km,
            const __bf16* __restrict__ Vm, const float* __restrict__ ln_g,
            const float* __restrict__ ln_b, float* __restrict__ outF,
            __bf16* __restrict__ outB) {
  __shared__ __attribute__((aligned(16))) __bf16 qs[48][64];
  __shared__ __attribute__((aligned(16))) __bf16 ks[48][64];
  __shared__ __attribute__((aligned(16))) __bf16 vt[64][64];  // vt[d][c]
  __shared__ __attribute__((aligned(16))) __bf16 pb[48][64];  // probs
  __shared__ float sc[48][48];
  __shared__ float ctx[48][64];
  __shared__ float red[128];

  const int n    = blockIdx.x;
  const int t    = threadIdx.x;
  const int lane = t & 31;
  const int wave = t >> 5;
  const int lm   = lane & 15;
  const int lh   = lane >> 4;

  // stage q/k rows: 48 rows x 8 chunks (8 halves each) = 384 chunks
#pragma unroll
  for (int it = 0; it < 3; ++it) {
    int c = t + 128 * it;          // 0..383
    int row = c >> 3, seg = (c & 7) * 8;
    if (row < COLS) {
      size_t base = ((size_t)n * COLS + row) * EMB + seg;
      async_cp16(&qs[row][seg], Q + base);
      async_cp16(&ks[row][seg], Km + base);
    } else {
      *(uint4*)&qs[row][seg] = (uint4){0u, 0u, 0u, 0u};
      *(uint4*)&ks[row][seg] = (uint4){0u, 0u, 0u, 0u};
    }
  }
  // stage v transposed (scalar; needs transpose anyway)
#pragma unroll
  for (int i = 0; i < 32; ++i) {
    int idx = t + 128 * i;
    int d = idx >> 6, c = idx & 63;
    __bf16 v = (__bf16)0.f;
    if (c < COLS) v = Vm[((size_t)n * COLS + c) * EMB + d];
    vt[d][c] = v;
  }
  async_wait0();
  __syncthreads();

  // scores: 3x3 tiles of 16x16, K = 64
  for (int tile = wave; tile < 9; tile += 4) {
    int ti = tile / 3, tj = tile % 3;
    v8f a = (v8f){0.f, 0.f, 0.f, 0.f, 0.f, 0.f, 0.f, 0.f};
#pragma unroll
    for (int kk = 0; kk < 64; kk += 32) {
      v16bf af  = ld_frag(&qs[ti * 16 + lm][0], kk, lh);
      v16bf bfm = ld_frag(&ks[tj * 16 + lm][0], kk, lh);
      a = wmma_bf16(af, bfm, a);
    }
#pragma unroll
    for (int r = 0; r < 8; ++r) sc[ti * 16 + 8 * lh + r][tj * 16 + lm] = a[r];
  }
  __syncthreads();

  // softmax over 40 cols; probs bf16, K-padded to 64 with zeros
  if (t < 48) {
    int row = t;
    if (row < COLS) {
      float mx = -1e30f;
      for (int c = 0; c < COLS; ++c) mx = fmaxf(mx, sc[row][c]);
      float s = 0.f;
      for (int c = 0; c < COLS; ++c) s += __expf(sc[row][c] - mx);
      float inv = 1.f / s;
      for (int c = 0; c < 64; ++c)
        pb[row][c] = (c < COLS) ? (__bf16)(__expf(sc[row][c] - mx) * inv)
                                : (__bf16)0.f;
    } else {
      for (int c = 0; c < 64; ++c) pb[row][c] = (__bf16)0.f;
    }
  }
  __syncthreads();

  // ctx = probs @ v : 3x4 tiles, K padded to 64
  for (int tile = wave; tile < 12; tile += 4) {
    int ti = tile >> 2, tj = tile & 3;
    v8f a = (v8f){0.f, 0.f, 0.f, 0.f, 0.f, 0.f, 0.f, 0.f};
#pragma unroll
    for (int kk = 0; kk < 64; kk += 32) {
      v16bf af  = ld_frag(&pb[ti * 16 + lm][0], kk, lh);
      v16bf bfm = ld_frag(&vt[tj * 16 + lm][0], kk, lh);
      a = wmma_bf16(af, bfm, a);
    }
#pragma unroll
    for (int r = 0; r < 8; ++r) ctx[ti * 16 + 8 * lh + r][tj * 16 + lm] = a[r];
  }
  __syncthreads();

  // LayerNorm over flattened 2560 (rows 0..39 x 64)
  float s1 = 0.f, s2 = 0.f;
  for (int idx = t; idx < 2560; idx += 128) {
    float x = ctx[idx >> 6][idx & 63];
    s1 += x; s2 += x * x;
  }
  red[t] = s1; __syncthreads();
  for (int o = 64; o > 0; o >>= 1) { if (t < o) red[t] += red[t + o]; __syncthreads(); }
  float mean = red[0] / 2560.f; __syncthreads();
  red[t] = s2; __syncthreads();
  for (int o = 64; o > 0; o >>= 1) { if (t < o) red[t] += red[t + o]; __syncthreads(); }
  float var = red[0] / 2560.f - mean * mean;
  float rstd = rsqrtf(var + 1e-6f);
  for (int idx = t; idx < 2560; idx += 128) {
    float x = (ctx[idx >> 6][idx & 63] - mean) * rstd * ln_g[idx] + ln_b[idx];
    outF[(size_t)n * DCAT + idx] = x;
    outB[(size_t)n * DCAT + idx] = (__bf16)x;
  }
}

// ---------------------------------------------------------------------------
// Value path (VALU; tiny): vq/vk/vv, softmax(vq*vk)*vv, LN(256).
// ---------------------------------------------------------------------------
__device__ __forceinline__ float blk_sum(float v, float* red, int t) {
  red[t] = v; __syncthreads();
  for (int o = 128; o > 0; o >>= 1) { if (t < o) red[t] += red[t + o]; __syncthreads(); }
  float r = red[0]; __syncthreads(); return r;
}
__device__ __forceinline__ float blk_max(float v, float* red, int t) {
  red[t] = v; __syncthreads();
  for (int o = 128; o > 0; o >>= 1) { if (t < o) red[t] = fmaxf(red[t], red[t + o]); __syncthreads(); }
  float r = red[0]; __syncthreads(); return r;
}

__global__ void __launch_bounds__(256)
value_kernel(const float* __restrict__ val, const float* __restrict__ qw,
             const float* __restrict__ qb, const float* __restrict__ kw,
             const float* __restrict__ kb, const float* __restrict__ vw,
             const float* __restrict__ vb, const float* __restrict__ g,
             const float* __restrict__ bln, float* __restrict__ outF,
             __bf16* __restrict__ outB) {
  __shared__ float x[64];
  __shared__ float red[256];
  const int n = blockIdx.x, t = threadIdx.x;
  if (t < 64) x[t] = val[(size_t)n * VALS + t];
  __syncthreads();
  float a1 = 0.f, a2 = 0.f, a3 = 0.f;
  for (int i = 0; i < 64; ++i) {
    float xi = x[i];
    a1 += xi * qw[i * 256 + t];
    a2 += xi * kw[i * 256 + t];
    a3 += xi * vw[i * 256 + t];
  }
  float vq = a1 + qb[t], vk = a2 + kb[t], vv = a3 + vb[t];
  float s = vq * vk;
  float mx = blk_max(s, red, t);
  float e = __expf(s - mx);
  float se = blk_sum(e, red, t);
  float vctx = e / se * vv;
  float mean = blk_sum(vctx, red, t) / 256.f;
  float var = blk_sum(vctx * vctx, red, t) / 256.f - mean * mean;
  float y = (vctx - mean) * rsqrtf(var + 1e-6f) * g[t] + bln[t];
  outF[(size_t)n * DCAT + 2560 + t] = y;
  outB[(size_t)n * DCAT + 2560 + t] = (__bf16)y;
}

// ---------------------------------------------------------------------------
// Small elementwise / segment kernels
// ---------------------------------------------------------------------------
__global__ void conv_bf16_kernel(const float* __restrict__ src,
                                 __bf16* __restrict__ dst, int n) {
  int i = blockIdx.x * blockDim.x + threadIdx.x;
  if (i < n) dst[i] = (__bf16)src[i];
}

__global__ void counts_kernel(const int* __restrict__ cl, float* __restrict__ cnt) {
  int i = blockIdx.x * blockDim.x + threadIdx.x;
  if (i < NROWS) atomicAdd(cnt + cl[i], 1.0f);
}

__global__ void __launch_bounds__(256)
segsum_kernel(const float* __restrict__ src, const int* __restrict__ cl,
              float* __restrict__ sums) {
  int n = blockIdx.x;
  int c = cl[n];
  for (int d = threadIdx.x; d < DCAT; d += 256)
    atomicAdd(sums + (size_t)c * DCAT + d, src[(size_t)n * DCAT + d]);
}

__global__ void means_kernel(const float* __restrict__ sums,
                             const float* __restrict__ cnt,
                             __bf16* __restrict__ means) {
  int i = blockIdx.x * blockDim.x + threadIdx.x;
  if (i < NC * DCAT) {
    float c = fmaxf(cnt[i / DCAT], 1.0f);
    means[i] = (__bf16)(sums[i] / c);
  }
}

__global__ void mix_kernel(const float* __restrict__ outF,
                           const float* __restrict__ gnnF,
                           const float* __restrict__ alpha,
                           __bf16* __restrict__ mixed, int n) {
  int i = blockIdx.x * blockDim.x + threadIdx.x;
  if (i < n) {
    float a = alpha[0];
    mixed[i] = (__bf16)(a * outF[i] + (1.0f - a) * gnnF[i]);
  }
}

__global__ void copyval_kernel(const float* __restrict__ val,
                               __bf16* __restrict__ out1) {
  int i = blockIdx.x * blockDim.x + threadIdx.x;
  if (i < NROWS * VALS) {
    int n = i / VALS, j = i % VALS;
    out1[(size_t)n * 576 + 512 + j] = (__bf16)val[i];
  }
}

__global__ void cat4_kernel(const float* __restrict__ out3,
                            const float* __restrict__ w,
                            const float* __restrict__ b,
                            float* __restrict__ out4) {
  int i = blockIdx.x * blockDim.x + threadIdx.x;
  if (i < NROWS * 2) {
    int n = i >> 1, j = i & 1;
    float s = b[j];
    for (int k = 0; k < 32; ++k) s += out3[(size_t)n * 32 + k] * w[k * 2 + j];
    out4[i] = fmaxf(s, 0.f);
  }
}

// ---------------------------------------------------------------------------
// Host launcher
// ---------------------------------------------------------------------------
extern "C" void kernel_launch(void* const* d_in, const int* in_sizes, int n_in,
                              void* d_out, int out_size, void* d_ws, size_t ws_size,
                              hipStream_t stream) {
  const float* value_batch = (const float*)d_in[0];
  const int*   embedd      = (const int*)d_in[1];
  const int*   clusters    = (const int*)d_in[2];
  const float* emb_table   = (const float*)d_in[3];
  const float* q_w = (const float*)d_in[4],  * q_b = (const float*)d_in[5];
  const float* k_w = (const float*)d_in[6],  * k_b = (const float*)d_in[7];
  const float* v_w = (const float*)d_in[8],  * v_b = (const float*)d_in[9];
  const float* ln_att_g = (const float*)d_in[10], * ln_att_b = (const float*)d_in[11];
  const float* qv_w = (const float*)d_in[12], * qv_b = (const float*)d_in[13];
  const float* kv_w = (const float*)d_in[14], * kv_b = (const float*)d_in[15];
  const float* vv_w = (const float*)d_in[16], * vv_b = (const float*)d_in[17];
  const float* ln_val_g = (const float*)d_in[18], * ln_val_b = (const float*)d_in[19];
  const float* gnn1_w = (const float*)d_in[20], * gnn1_b = (const float*)d_in[21];
  const float* gnn2_w = (const float*)d_in[22], * gnn2_b = (const float*)d_in[23];
  const float* cat1_w = (const float*)d_in[24], * cat1_b = (const float*)d_in[25];
  const float* cat2_w = (const float*)d_in[26], * cat2_b = (const float*)d_in[27];
  const float* cat3_w = (const float*)d_in[28], * cat3_b = (const float*)d_in[29];
  const float* cat4_w = (const float*)d_in[30], * cat4_b = (const float*)d_in[31];
  const float* dec1_w = (const float*)d_in[32], * dec1_b = (const float*)d_in[33];
  const float* dec2_w = (const float*)d_in[34], * dec2_b = (const float*)d_in[35];
  const float* alpha  = (const float*)d_in[36];

  // workspace layout
  char* p = (char*)d_ws;
  auto alloc = [&](size_t bytes) -> char* {
    char* r = p;
    p += (bytes + 255) & ~(size_t)255;
    return r;
  };
  float*  outF   = (float*)alloc((size_t)NROWS * DCAT * 4);
  __bf16* outB   = (__bf16*)alloc((size_t)NROWS * DCAT * 2);
  __bf16* embB   = (__bf16*)alloc((size_t)10000 * EMB * 2);
  __bf16* qB     = (__bf16*)alloc((size_t)NROWS * COLS * EMB * 2);
  __bf16* kB     = (__bf16*)alloc((size_t)NROWS * COLS * EMB * 2);
  __bf16* vB     = (__bf16*)alloc((size_t)NROWS * COLS * EMB * 2);
  float*  sums   = (float*)alloc((size_t)NC * DCAT * 4);
  float*  cnts   = (float*)alloc(512);
  __bf16* meansB = (__bf16*)alloc((size_t)NC * DCAT * 2);
  float*  gnnF   = (float*)alloc((size_t)NROWS * DCAT * 4);
  __bf16* mixedB = (__bf16*)alloc((size_t)NROWS * DCAT * 2);
  __bf16* out1B  = (__bf16*)alloc((size_t)NROWS * 576 * 2);
  __bf16* out2B  = (__bf16*)alloc((size_t)NROWS * 128 * 2);
  float*  out3F  = (float*)alloc((size_t)NROWS * 32 * 4);
  __bf16* dec1B  = (__bf16*)alloc((size_t)NROWS * 1024 * 2);
  // transposed bf16 weights Wt[N][K]
  __bf16* wtq    = (__bf16*)alloc((size_t)EMB * EMB * 2);
  __bf16* wtk    = (__bf16*)alloc((size_t)EMB * EMB * 2);
  __bf16* wtv    = (__bf16*)alloc((size_t)EMB * EMB * 2);
  __bf16* wtg1   = (__bf16*)alloc((size_t)DCAT * DCAT * 2);
  __bf16* wtg2   = (__bf16*)alloc((size_t)DCAT * DCAT * 2);
  __bf16* wtc1   = (__bf16*)alloc((size_t)DCAT * 512 * 2);
  __bf16* wtc2   = (__bf16*)alloc((size_t)576 * 128 * 2);
  __bf16* wtc3   = (__bf16*)alloc((size_t)128 * 32 * 2);
  __bf16* wtd1   = (__bf16*)alloc((size_t)DCAT * 1024 * 2);
  __bf16* wtd2   = (__bf16*)alloc((size_t)1024 * 104 * 2);
  // gnn1 f32 activations alias the (by-then dead) q/k/v bf16 region
  float*  gnn1F  = (float*)qB;

  // output regions: (out4 [4096,2], dec2 [4096,104], out2 [4096,128])
  float* out4_d = (float*)d_out;
  float* dec2_d = (float*)d_out + (size_t)NROWS * 2;
  float* out2_d = (float*)d_out + (size_t)NROWS * 2 + (size_t)NROWS * 104;

  auto twt = [&](const float* W, __bf16* Wt, int K, int N) {
    int n = K * N;
    twt_kernel<<<(n + 255) / 256, 256, 0, stream>>>(W, Wt, K, N);
  };
  auto gemm = [&](const __bf16* A, int lda, const int* rmap, const __bf16* Wt,
                  const float* bias, int M, int N, int K, int relu, float* Cf,
                  __bf16* Cb, int ldc) {
    dim3 grid((N + BN - 1) / BN, (M + BM - 1) / BM);
    gemm_bf16_kernel<<<grid, 256, 0, stream>>>(A, lda, rmap, Wt, bias, M, N, K,
                                               relu, Cf, Cb, ldc);
  };

  // 0) weight prep (bf16 transpose) + embedding table to bf16
  twt(q_w, wtq, EMB, EMB);
  twt(k_w, wtk, EMB, EMB);
  twt(v_w, wtv, EMB, EMB);
  twt(gnn1_w, wtg1, DCAT, DCAT);
  twt(gnn2_w, wtg2, DCAT, DCAT);
  twt(cat1_w, wtc1, DCAT, 512);
  twt(cat2_w, wtc2, 576, 128);
  twt(cat3_w, wtc3, 128, 32);
  twt(dec1_w, wtd1, DCAT, 1024);
  twt(dec2_w, wtd2, 1024, 104);
  {
    int n = 10000 * EMB;
    conv_bf16_kernel<<<(n + 255) / 256, 256, 0, stream>>>(emb_table, embB, n);
  }
  // 1) value path -> output cols [2560,2816)
  value_kernel<<<NROWS, 256, 0, stream>>>(value_batch, qv_w, qv_b, kv_w, kv_b,
                                          vv_w, vv_b, ln_val_g, ln_val_b, outF, outB);
  // 2) q/k/v projections via gather-GEMM (M = 4096*40)
  const int MQ = NROWS * COLS;
  gemm(embB, EMB, embedd, wtq, q_b, MQ, EMB, EMB, 0, nullptr, qB, EMB);
  gemm(embB, EMB, embedd, wtk, k_b, MQ, EMB, EMB, 0, nullptr, kB, EMB);
  gemm(embB, EMB, embedd, wtv, v_b, MQ, EMB, EMB, 0, nullptr, vB, EMB);
  // 3) attention + LN -> output cols [0,2560)
  attn_kernel<<<NROWS, 128, 0, stream>>>(qB, kB, vB, ln_att_g, ln_att_b, outF, outB);

  // 4) cluster mean #1
  hipMemsetAsync(sums, 0, (size_t)NC * DCAT * 4, stream);
  hipMemsetAsync(cnts, 0, 512, stream);
  counts_kernel<<<(NROWS + 255) / 256, 256, 0, stream>>>(clusters, cnts);
  segsum_kernel<<<NROWS, 256, 0, stream>>>(outF, clusters, sums);
  means_kernel<<<(NC * DCAT + 255) / 256, 256, 0, stream>>>(sums, cnts, meansB);
  // 5) gnn1 = relu(cm(output) @ W1 + b1)
  gemm(meansB, DCAT, clusters, wtg1, gnn1_b, NROWS, DCAT, DCAT, 1, gnn1F,
       nullptr, DCAT);
  // 6) cluster mean #2
  hipMemsetAsync(sums, 0, (size_t)NC * DCAT * 4, stream);
  segsum_kernel<<<NROWS, 256, 0, stream>>>(gnn1F, clusters, sums);
  means_kernel<<<(NC * DCAT + 255) / 256, 256, 0, stream>>>(sums, cnts, meansB);
  // 7) gnn = relu(cm(gnn1) @ W2 + b2)
  gemm(meansB, DCAT, clusters, wtg2, gnn2_b, NROWS, DCAT, DCAT, 1, gnnF,
       nullptr, DCAT);
  // 8) mixed = alpha*output + (1-alpha)*gnn
  {
    int n = NROWS * DCAT;
    mix_kernel<<<(n + 255) / 256, 256, 0, stream>>>(outF, gnnF, alpha, mixedB, n);
  }
  // 9) cat1 -> out1 cols [0,512); concat value cols [512,576)
  gemm(mixedB, DCAT, nullptr, wtc1, cat1_b, NROWS, 512, DCAT, 1, nullptr,
       out1B, 576);
  copyval_kernel<<<(NROWS * VALS + 255) / 256, 256, 0, stream>>>(value_batch, out1B);
  // 10) out2 = relu(out1 @ cat2)  (f32 straight into d_out + bf16 copy)
  gemm(out1B, 576, nullptr, wtc2, cat2_b, NROWS, 128, 576, 1, out2_d, out2B, 128);
  // 11) out3 = relu(out2 @ cat3)
  gemm(out2B, 128, nullptr, wtc3, cat3_b, NROWS, 32, 128, 1, out3F, nullptr, 32);
  // 12) out4 = relu(out3 @ cat4)  (N=2: VALU)
  cat4_kernel<<<(NROWS * 2 + 255) / 256, 256, 0, stream>>>(out3F, cat4_w, cat4_b,
                                                           out4_d);
  // 13) dec1 = relu(output @ dec1_w)
  gemm(outB, DCAT, nullptr, wtd1, dec1_b, NROWS, 1024, DCAT, 1, nullptr,
       dec1B, 1024);
  // 14) dec2 = relu(dec1 @ dec2_w) -> d_out
  gemm(dec1B, 1024, nullptr, wtd2, dec2_b, NROWS, 104, 1024, 1, dec2_d,
       nullptr, 104);

  (void)in_sizes; (void)n_in; (void)out_size; (void)ws_size;
}
